// Neuralsymbol_64433099375164
// MI455X (gfx1250) — matrix operand order repaired and checked by reading, not compiled, not run on hardware
//
#include <hip/hip_runtime.h>
#include <hip/hip_bf16.h>

// ---------------------------------------------------------------------------
// Neural-symbol stack machine, persistent-workgroup WMMA implementation.
//   grid = 8 workgroups (16 batch rows each), block = 128 threads (4 waves).
//   Stacks in VGPRs; per-step vectors in one linear per-row LDS buffer;
//   weights row-normalized + f16 fragment-packed in d_ws (L2 resident).
//   Small GEMMs are split across all 4 waves in N and K; K-partials are
//   combined with LDS ds_add_f32 atomics.
// ---------------------------------------------------------------------------

typedef _Float16 v16h __attribute__((ext_vector_type(16)));
typedef float    v8f  __attribute__((ext_vector_type(8)));

#define DEV __device__ __forceinline__

#define ZO   1e-6f
#define EPS_ 1e-8f

// per-row vector layout inside s_vec (stride VSTR floats per row)
#define SEG_CVAL 0
#define SEG_WVAL 1
#define SEG_CPK  2
#define SEG_WPK  3
#define SEG_ITYP 4
#define SEG_INP  5
#define VSTR 772          // 6*128 + 4 pad (de-conflicts LDS banks across rows)
#define LSTR 81           // logits row stride

// packed weight layout (halfs), fragment-ready order [group][nt][kt][lane][16]
#define OFF_SYM   0        //  1 nt x  4 kt
#define OFF_PRE   2048     //  2 nt x 12 kt
#define OFF_POST  14336    //  5 nt x 16 kt
#define OFF_PW    55296    //  5 nt x  8 kt
#define OFF_SEL   75776    //  2 nt x 20 kt
#define PACK_HALFS 96256
#define NORM_BYTE_OFF (PACK_HALFS * 2)   // 400 floats of row rnorms follow

struct WPtrs {
  const float *Wsym, *Wpcp, *Wpcn, *Wpwp, *Wpwn;
  const float *Wcpush, *Wcnop, *Wcch, *Wwpush, *Wwnop;
  const float *Wwch, *Wsel;
};

__device__ __constant__ const int kDg[5] = {128, 384, 512, 256, 640};
__device__ __constant__ const int kRg[5] = {12, 32, 80, 80, 24};

DEV const float* row_src(const WPtrs& w, int g, int row) {
  switch (g) {
    case 0: return w.Wsym + row * 128;
    case 1: {
      int q = row >> 3, r = row & 7;
      const float* b = (q == 0) ? w.Wpcp : (q == 1) ? w.Wpcn : (q == 2) ? w.Wpwp : w.Wpwn;
      return b + r * 384;
    }
    case 2:
      if (row < 8)  return w.Wcpush + row * 512;
      if (row < 16) return w.Wcnop + (row - 8) * 512;
      if (row < 64) return w.Wcch + (row - 16) * 512;
      if (row < 72) return w.Wwpush + (row - 64) * 512;
      return w.Wwnop + (row - 72) * 512;
    case 3: return w.Wwch + row * 256;
    default: return w.Wsel + row * 640;
  }
}

// ------------------------ prep: per-row L2 norms ---------------------------
__global__ void prep_norms(WPtrs w, float* __restrict__ norms) {
  int id = blockIdx.x * blockDim.x + threadIdx.x;
  if (id >= 400) return;
  int g = id / 80, row = id % 80;
  if (row >= kRg[g]) { norms[id] = 0.f; return; }
  const float* src = row_src(w, g, row);
  int D = kDg[g];
  float s = 0.f;
  for (int k = 0; k < D; ++k) { float v = src[k]; s += v * v; }
  norms[id] = rsqrtf(s + EPS_);
}

// ------------- prep: pack normalized f16 B fragments into d_ws -------------
// B fragment (f16, 32xK-tile x 16): lane<16 -> n=lane,    elems 0..7 K=0..7,  8..15 K=16..23
//                                   lane>=16-> n=lane-16, elems 0..7 K=8..15, 8..15 K=24..31
__global__ void prep_pack(WPtrs w, const float* __restrict__ norms,
                          _Float16* __restrict__ pack) {
  int idx = blockIdx.x * blockDim.x + threadIdx.x;
  if (idx >= PACK_HALFS) return;
  int g, off, KT;
  if (idx < OFF_PRE)       { g = 0; off = OFF_SYM;  KT = 4;  }
  else if (idx < OFF_POST) { g = 1; off = OFF_PRE;  KT = 12; }
  else if (idx < OFF_PW)   { g = 2; off = OFF_POST; KT = 16; }
  else if (idx < OFF_SEL)  { g = 3; off = OFF_PW;   KT = 8;  }
  else                     { g = 4; off = OFF_SEL;  KT = 20; }
  int rel  = idx - off;
  int tile = rel >> 9;            // 512 halfs per 32x16 tile
  int lane = (rel >> 4) & 31;
  int i    = rel & 15;
  int nt = tile / KT, kt = tile - nt * KT;
  int row = nt * 16 + (lane & 15);
  int k   = kt * 32 + ((lane & 16) ? 8 : 0) + ((i < 8) ? i : (8 + i));
  float v = 0.f;
  if (row < kRg[g]) v = row_src(w, g, row)[k] * norms[g * 80 + row];
  pack[idx] = (_Float16)v;
}

// ----------------------------- GEMM helper ---------------------------------
// A (16xK f16) gathered linearly from s_vec (concats are contiguous), scaled
// by per-row rnorm during f16 convert.
// A layout: lane<16 -> m=lane,    elems 0..7 K=kb..kb+7,  8..15 K=kb+16..kb+23
//           lane>=16-> m=lane-16, kb += 8
// Work units = NT x KSPLIT spread over 4 waves; KSPLIT>1 combines K-partials
// with LDS float atomics into a pre-zeroed logits tile.
template <int KT, int NT, int KSPLIT, int SEGOFF, bool PW_GAP>
DEV void gemm_phase(const _Float16* __restrict__ bpack,
                    float* __restrict__ logits,
                    const float* __restrict__ svec,
                    float rn, int lane, int wave) {
  constexpr int KC = KT / KSPLIT;
  const int m = lane & 15;
  const float* sp = svec + m * VSTR + SEGOFF * 128;
  for (int u = wave; u < NT * KSPLIT; u += 4) {
    const int nt = u / KSPLIT, kc = u % KSPLIT;
    const float* spc = sp + kc * (KC * 32);
    v8f acc = {};
#pragma unroll
    for (int kk = 0; kk < KC; ++kk) {
      const int kb = kk * 32 + ((lane & 16) ? 8 : 0);
      const int adj = PW_GAP ? ((kk * 32) & ~127) : 0;  // jump wpk -> inp region
      v16h a;
#pragma unroll
      for (int i = 0; i < 16; ++i) {
        int k = kb + ((i < 8) ? i : (8 + i));
        a[i] = (_Float16)(spc[k + adj] * rn);
      }
      v16h b = *(const v16h*)(bpack + (size_t)((nt * KT + kc * KC + kk) * 32 + lane) * 16);
      acc = __builtin_amdgcn_wmma_f32_16x16x32_f16(false, a, false, b,
                                                   (short)0, acc, false, false);
    }
    int col = nt * 16 + m;
    int rb  = (lane & 16) ? 8 : 0;
#pragma unroll
    for (int r = 0; r < 8; ++r) {
      if (KSPLIT > 1) atomicAdd(&logits[(rb + r) * LSTR + col], acc[r]);  // ds_add_f32
      else            logits[(rb + r) * LSTR + col] = acc[r];
    }
  }
}

// ------------------------- scalar helpers ----------------------------------
DEV void choice_probs(const float* lg, int CR, float scale, float* pOut) {
  float mx = -1e30f;
  for (int i = 0; i < CR; ++i) mx = fmaxf(mx, lg[i] * scale);
  float s = 0.f;
  for (int i = 0; i < CR; ++i) s += __expf(lg[i] * scale - mx);
  float inv = 1.f / s;
  for (int c = 0; c < (CR >> 2); ++c) {
    float t = 0.f;
    for (int r = 0; r < 4; ++r) t += __expf(lg[c * 4 + r] * scale - mx);
    pOut[c] = t * inv;
  }
}

DEV float op_gate(const float* lg, float scale) {
  float mx = -1e30f;
#pragma unroll
  for (int i = 0; i < 8; ++i) mx = fmaxf(mx, lg[i] * scale);
  float e[8], s = 0.f;
#pragma unroll
  for (int i = 0; i < 8; ++i) { e[i] = __expf(lg[i] * scale - mx); s += e[i]; }
  float inv = 1.f / s;
  float c0 = (e[0] + e[1] + e[2] + e[3]) * inv;
  float c1 = (e[4] + e[5] + e[6] + e[7]) * inv;
  return (c0 - c1 + 1.f) * 0.5f;
}

DEV void mix2(float pd, float pn, float sharp, float& a0, float& a1) {
  float x0 = pd * sharp, x1 = pn * sharp, m = fmaxf(x0, x1);
  float e0 = __expf(x0 - m), e1 = __expf(x1 - m), inv = 1.f / (e0 + e1);
  a0 = e0 * inv; a1 = e1 * inv;
}

// ------------------------------ main kernel --------------------------------
__global__ __launch_bounds__(128) void ns_main(
    const float* __restrict__ x,           // [128,512,128]
    const float* __restrict__ sym_outs,    // [128,3]
    const float* __restrict__ pc_outs,     // [128,12]
    const float* __restrict__ pw_outs,     // [128,20]
    const float* __restrict__ sel_outs,    // [128,5]
    const float* __restrict__ init_vec,    // [128]
    const float* __restrict__ ctrl_sharp_p,
    const float* __restrict__ work_sharp_p,
    const float* __restrict__ scales,      // [12]
    const _Float16* __restrict__ pack,
    float* __restrict__ out) {
  const int tid  = threadIdx.x;
  const int lane = tid & 31, wave = tid >> 5;
  const int row  = tid >> 3;          // stack-owner row, 8 threads per row
  const int vc   = (tid & 7) * 16;    // 16 v-slots per thread
  const int bBase = blockIdx.x * 16;

  __shared__ float s_vec[16 * VSTR];      // [row][cval|wval|cpk|wpk|ityp|inp]
  __shared__ float s_logits[16 * LSTR];
  __shared__ float s_part[6 * 128];       // per-thread sum-of-squares partials
  __shared__ float s_psym[16 * 3], s_pctl[16 * 12], s_pwrk[16 * 20], s_sel[16 * 6];
  __shared__ float s_cptr[16 * 4], s_wptr[16 * 4];
  __shared__ float s_gate[16 * 8];        // a-pairs: pre_c, pre_w, post_c, post_w
  __shared__ float s_rn[16 * 5];          // rn_inp, rn_pre, rn_post, rn_pw, rn_sel

  const float ctrl_sharp = *ctrl_sharp_p;
  const float work_sharp = *work_sharp_p;
  float cs[12];
#pragma unroll
  for (int i = 0; i < 12; ++i) cs[i] = scales[i];

  float* myrow = &s_vec[row * VSTR];

  // Register-resident stacks: [slot][my 16 v entries]
  float cst[4][16], wst[4][16];
#pragma unroll
  for (int n = 0; n < 4; ++n)
#pragma unroll
    for (int j = 0; j < 16; ++j) {
      float v = (n == 1) ? init_vec[vc + j] : ZO;
      cst[n][j] = v; wst[n][j] = v;
    }
  if (tid < 16) {
#pragma unroll
    for (int n = 0; n < 4; ++n) {
      float p = (n == 1) ? 1.f : 0.f;
      s_cptr[tid * 4 + n] = p; s_wptr[tid * 4 + n] = p;
    }
  }
  __syncthreads();

  const _Float16* packSym  = pack + OFF_SYM;
  const _Float16* packPre  = pack + OFF_PRE;
  const _Float16* packPost = pack + OFF_POST;
  const _Float16* packPw   = pack + OFF_PW;
  const _Float16* packSel  = pack + OFF_SEL;

  for (int t = 0; t < 512; ++t) {
    // ---- load x tile, accumulate |inp|^2 partial, zero logits -----------
    {
      const float* src = x + ((size_t)(bBase + row) * 512 + t) * 128 + vc;
      float sx = 0.f;
#pragma unroll
      for (int j = 0; j < 16; j += 4) {
        float4 f = *(const float4*)(src + j);
        *(float4*)&myrow[SEG_INP * 128 + vc + j] = f;
        sx += f.x * f.x + f.y * f.y + f.z * f.z + f.w * f.w;
      }
      s_part[3 * 128 + tid] = sx;
      if (t + 1 < 512) __builtin_prefetch(src + 128, 0, 1);
      for (int i = tid; i < 16 * LSTR; i += 128) s_logits[i] = 0.f;  // for GEMM1
    }
    __syncthreads();
    if (tid < 16) {               // rn for [inp]
      float s = 0.f;
#pragma unroll
      for (int j = 0; j < 8; ++j) s += s_part[3 * 128 + tid * 8 + j];
      s_rn[0 * 16 + tid] = rsqrtf(s + EPS_);
    }
    __syncthreads();

    // ---- GEMM1: sym_type logits, K split over all 4 waves ---------------
    gemm_phase<4, 1, 4, SEG_INP, false>(packSym, s_logits, s_vec,
                                        s_rn[0 * 16 + (lane & 15)], lane, wave);
    __syncthreads();
    if (tid < 16) choice_probs(&s_logits[tid * LSTR], 12, cs[0], &s_psym[tid * 3]);
    __syncthreads();

    // ---- inp_typ + peeks (+ sumsq partials), zero logits for GEMM2 ------
    {
      float p0 = s_psym[row * 3 + 0], p1 = s_psym[row * 3 + 1], p2 = s_psym[row * 3 + 2];
      float sc = 0.f, sw = 0.f, si = 0.f;
#pragma unroll
      for (int j = 0; j < 16; ++j) {
        int v = vc + j;
        float it = p0 * sym_outs[v * 3 + 0] + p1 * sym_outs[v * 3 + 1] +
                   p2 * sym_outs[v * 3 + 2];
        myrow[SEG_ITYP * 128 + v] = it; si += it * it;
        float cp = 0.f, wp = 0.f;
#pragma unroll
        for (int n = 0; n < 4; ++n) {
          cp += cst[n][j] * s_cptr[row * 4 + n];
          wp += wst[n][j] * s_wptr[row * 4 + n];
        }
        myrow[SEG_CPK * 128 + v] = cp; sc += cp * cp;
        myrow[SEG_WPK * 128 + v] = wp; sw += wp * wp;
      }
      s_part[0 * 128 + tid] = sc;
      s_part[1 * 128 + tid] = sw;
      s_part[2 * 128 + tid] = si;
      for (int i = tid; i < 16 * LSTR; i += 128) s_logits[i] = 0.f;
    }
    __syncthreads();
    if (tid < 16) {               // rn for pre/post/pw concats
      float sc = 0.f, sw = 0.f, si = 0.f, sx = 0.f;
#pragma unroll
      for (int j = 0; j < 8; ++j) {
        int p = tid * 8 + j;
        sc += s_part[0 * 128 + p]; sw += s_part[1 * 128 + p];
        si += s_part[2 * 128 + p]; sx += s_part[3 * 128 + p];
      }
      s_rn[1 * 16 + tid] = rsqrtf(sc + sw + si + EPS_);
      s_rn[2 * 16 + tid] = rsqrtf(sc + sw + si + sx + EPS_);
      s_rn[3 * 16 + tid] = rsqrtf(sw + sx + EPS_);
    }
    __syncthreads();

    // ---- GEMM2: packed pre ops [pcp|pcn|pwp|pwn], D=384, K split 2 ------
    gemm_phase<12, 2, 2, SEG_CPK, false>(packPre, s_logits, s_vec,
                                         s_rn[1 * 16 + (lane & 15)], lane, wave);
    __syncthreads();
    if (tid < 16) {
      const float* lg = &s_logits[tid * LSTR];
      float a0, a1;
      mix2(op_gate(lg + 0, cs[1]),  op_gate(lg + 8, cs[2]),  ctrl_sharp, a0, a1);
      s_gate[tid * 8 + 0] = a0; s_gate[tid * 8 + 1] = a1;
      mix2(op_gate(lg + 16, cs[3]), op_gate(lg + 24, cs[4]), work_sharp, a0, a1);
      s_gate[tid * 8 + 2] = a0; s_gate[tid * 8 + 3] = a1;
    }
    __syncthreads();

    // ---- soft pop_or_nop on both stacks ---------------------------------
    {
      float pc[4], pw[4];
#pragma unroll
      for (int n = 0; n < 4; ++n) { pc[n] = s_cptr[row * 4 + n]; pw[n] = s_wptr[row * 4 + n]; }
      float ca0 = s_gate[row * 8 + 0], ca1 = s_gate[row * 8 + 1];
      float wa0 = s_gate[row * 8 + 2], wa1 = s_gate[row * 8 + 3];
#pragma unroll
      for (int n = 0; n < 4; ++n)
#pragma unroll
        for (int j = 0; j < 16; ++j) {
          float v = cst[n][j]; cst[n][j] = ca0 * (v * (1.f - pc[n]) + pc[n] * ZO) + ca1 * v;
          v = wst[n][j];       wst[n][j] = wa0 * (v * (1.f - pw[n]) + pw[n] * ZO) + wa1 * v;
        }
      float npc[4], npw[4];
      if (tid < 16) {
        float qc[4], qw[4];
#pragma unroll
        for (int n = 0; n < 4; ++n) { qc[n] = s_cptr[tid * 4 + n]; qw[n] = s_wptr[tid * 4 + n]; }
        float ka0 = s_gate[tid * 8 + 0], ka1 = s_gate[tid * 8 + 1];
        float ja0 = s_gate[tid * 8 + 2], ja1 = s_gate[tid * 8 + 3];
#pragma unroll
        for (int n = 0; n < 4; ++n) {
          npc[n] = ka0 * qc[(n + 1) & 3] + ka1 * qc[n];   // roll(ptr,-1) mixed
          npw[n] = ja0 * qw[(n + 1) & 3] + ja1 * qw[n];
        }
      }
      __syncthreads();
      if (tid < 16)
#pragma unroll
        for (int n = 0; n < 4; ++n) { s_cptr[tid * 4 + n] = npc[n]; s_wptr[tid * 4 + n] = npw[n]; }
    }
    __syncthreads();

    // ---- GEMM3: packed post [c_push|c_nop|c_choice48|w_push|w_nop], D=512
    gemm_phase<16, 5, 1, SEG_CPK, false>(packPost, s_logits, s_vec,
                                         s_rn[2 * 16 + (lane & 15)], lane, wave);
    __syncthreads();
    if (tid < 16) {
      const float* lg = &s_logits[tid * LSTR];
      choice_probs(lg + 16, 48, cs[7], &s_pctl[tid * 12]);
      float a0, a1;
      mix2(op_gate(lg + 0, cs[5]),  op_gate(lg + 8, cs[6]),  ctrl_sharp, a0, a1);
      s_gate[tid * 8 + 4] = a0; s_gate[tid * 8 + 5] = a1;
      mix2(op_gate(lg + 64, cs[8]), op_gate(lg + 72, cs[9]), work_sharp, a0, a1);
      s_gate[tid * 8 + 6] = a0; s_gate[tid * 8 + 7] = a1;
    }
    __syncthreads();
    {  // control_val = p_ctl @ post_c_outputs^T  (+ sumsq partial)
      float scv = 0.f;
#pragma unroll
      for (int j = 0; j < 16; ++j) {
        int v = vc + j; float acc = 0.f;
#pragma unroll
        for (int c = 0; c < 12; ++c) acc += s_pctl[row * 12 + c] * pc_outs[v * 12 + c];
        myrow[SEG_CVAL * 128 + v] = acc; scv += acc * acc;
      }
      s_part[4 * 128 + tid] = scv;
    }
    __syncthreads();

    // ---- GEMM4: post_w_choice [80,256] over [w_peek, inp] ---------------
    gemm_phase<8, 5, 1, SEG_WPK, true>(packPw, s_logits, s_vec,
                                       s_rn[3 * 16 + (lane & 15)], lane, wave);
    __syncthreads();
    if (tid < 16) choice_probs(&s_logits[tid * LSTR], 80, cs[10], &s_pwrk[tid * 20]);
    __syncthreads();
    {  // work_val = p_wrk @ post_w_outputs^T (+ partial), zero logits for GEMM5
      float swv = 0.f;
#pragma unroll
      for (int j = 0; j < 16; ++j) {
        int v = vc + j; float acc = 0.f;
        for (int c = 0; c < 20; ++c) acc += s_pwrk[row * 20 + c] * pw_outs[v * 20 + c];
        myrow[SEG_WVAL * 128 + v] = acc; swv += acc * acc;
      }
      s_part[5 * 128 + tid] = swv;
      for (int i = tid; i < 16 * LSTR; i += 128) s_logits[i] = 0.f;
    }
    __syncthreads();

    // ---- soft push_or_nop on both stacks, + rn_sel ----------------------
    {
      float pc[4], pw[4];
#pragma unroll
      for (int n = 0; n < 4; ++n) { pc[n] = s_cptr[row * 4 + n]; pw[n] = s_wptr[row * 4 + n]; }
      float ca0 = s_gate[row * 8 + 4], ca1 = s_gate[row * 8 + 5];
      float wa0 = s_gate[row * 8 + 6], wa1 = s_gate[row * 8 + 7];
#pragma unroll
      for (int n = 0; n < 4; ++n) {
        float prc = pc[(n + 3) & 3], prw = pw[(n + 3) & 3];   // roll(ptr,+1)
#pragma unroll
        for (int j = 0; j < 16; ++j) {
          float cv = myrow[SEG_CVAL * 128 + vc + j], wv = myrow[SEG_WVAL * 128 + vc + j];
          float v = cst[n][j]; cst[n][j] = ca0 * (v * (1.f - prc) + prc * cv) + ca1 * v;
          v = wst[n][j];       wst[n][j] = wa0 * (v * (1.f - prw) + prw * wv) + wa1 * v;
        }
      }
      float npc[4], npw[4], rnsel = 0.f;
      if (tid < 16) {
        float qc[4], qw[4];
#pragma unroll
        for (int n = 0; n < 4; ++n) { qc[n] = s_cptr[tid * 4 + n]; qw[n] = s_wptr[tid * 4 + n]; }
        float ka0 = s_gate[tid * 8 + 4], ka1 = s_gate[tid * 8 + 5];
        float ja0 = s_gate[tid * 8 + 6], ja1 = s_gate[tid * 8 + 7];
#pragma unroll
        for (int n = 0; n < 4; ++n) {
          npc[n] = ka0 * qc[(n + 3) & 3] + ka1 * qc[n];
          npw[n] = ja0 * qw[(n + 3) & 3] + ja1 * qw[n];
        }
        float s = 0.f;
#pragma unroll
        for (int j = 0; j < 8; ++j) {
          int p = tid * 8 + j;
          s += s_part[0 * 128 + p] + s_part[1 * 128 + p] + s_part[2 * 128 + p] +
               s_part[4 * 128 + p] + s_part[5 * 128 + p];
        }
        rnsel = rsqrtf(s + EPS_);
      }
      __syncthreads();
      if (tid < 16) {
#pragma unroll
        for (int n = 0; n < 4; ++n) { s_cptr[tid * 4 + n] = npc[n]; s_wptr[tid * 4 + n] = npw[n]; }
        s_rn[4 * 16 + tid] = rnsel;
      }
    }
    __syncthreads();

    // ---- GEMM5: select [24(pad32), 640], K split 2 ----------------------
    gemm_phase<20, 2, 2, SEG_CVAL, false>(packSel, s_logits, s_vec,
                                          s_rn[4 * 16 + (lane & 15)], lane, wave);
    __syncthreads();
    if (tid < 16) choice_probs(&s_logits[tid * LSTR], 24, cs[11], &s_sel[tid * 6]);
    __syncthreads();

    // ---- out = sel[:5] @ select_out_outputs^T + sel[5] * work_val -------
    {
      float* op = out + ((size_t)(bBase + row) * 512 + t) * 128 + vc;
#pragma unroll
      for (int j = 0; j < 16; ++j) {
        int v = vc + j;
        float acc = s_sel[row * 6 + 5] * myrow[SEG_WVAL * 128 + v];
#pragma unroll
        for (int c = 0; c < 5; ++c) acc += s_sel[row * 6 + c] * sel_outs[v * 5 + c];
        op[j] = acc;
      }
    }
    __syncthreads();
  }
}

// ----------------------------- host launch ---------------------------------
extern "C" void kernel_launch(void* const* d_in, const int* in_sizes, int n_in,
                              void* d_out, int out_size, void* d_ws, size_t ws_size,
                              hipStream_t stream) {
  (void)in_sizes; (void)n_in; (void)out_size;

  WPtrs w;
  w.Wsym   = (const float*)d_in[8];
  w.Wpcp   = (const float*)d_in[9];
  w.Wpcn   = (const float*)d_in[10];
  w.Wpwp   = (const float*)d_in[11];
  w.Wpwn   = (const float*)d_in[12];
  w.Wcpush = (const float*)d_in[13];
  w.Wcnop  = (const float*)d_in[14];
  w.Wcch   = (const float*)d_in[15];
  w.Wwpush = (const float*)d_in[16];
  w.Wwnop  = (const float*)d_in[17];
  w.Wwch   = (const float*)d_in[18];
  w.Wsel   = (const float*)d_in[19];

  _Float16* pack = (_Float16*)d_ws;                            // 192512 bytes
  float* norms   = (float*)((char*)d_ws + NORM_BYTE_OFF);      // + 1600 bytes
  (void)ws_size;  // requires ~194 KB of scratch

  prep_norms<<<2, 256, 0, stream>>>(w, norms);
  prep_pack<<<(PACK_HALFS + 255) / 256, 256, 0, stream>>>(w, norms, pack);

  ns_main<<<8, 128, 0, stream>>>(
      (const float*)d_in[0],   // x
      (const float*)d_in[1],   // sym_type_outs
      (const float*)d_in[2],   // post_c_outputs
      (const float*)d_in[3],   // post_w_outputs
      (const float*)d_in[4],   // select_out_outputs
      (const float*)d_in[5],   // stack_init_vec
      (const float*)d_in[6],   // control_sharp
      (const float*)d_in[7],   // work_sharp
      (const float*)d_in[20],  // choice_scales
      pack,
      (float*)d_out);
}
